// ControlNet_Arch_Wrapper_72756745994308
// MI455X (gfx1250) — compile-verified, matrix operand order checked by reading, not run on hardware
//
#include <hip/hip_runtime.h>

// ---------------------------------------------------------------------------
// Types / helpers
// ---------------------------------------------------------------------------
typedef __attribute__((ext_vector_type(16))) __bf16        v16bf;
typedef __attribute__((ext_vector_type(8)))  float         v8f;
typedef __attribute__((ext_vector_type(8)))  unsigned int  v8u;

static constexpr int  NN = 32768;
static constexpr int  EE = 262144;
static constexpr int  LL = 4;

#define DEVFN __device__ __forceinline__

DEVFN unsigned short f2bf(float f) {
  union { float f; unsigned u; } v; v.f = f;
  unsigned r = v.u + 0x7FFFu + ((v.u >> 16) & 1u);   // round-to-nearest-even
  return (unsigned short)(r >> 16);
}
DEVFN unsigned pack2bf(float a, float b) {
  return (unsigned)f2bf(a) | ((unsigned)f2bf(b) << 16);
}
DEVFN float silu_f(float x) { return x * (1.0f / (1.0f + __expf(-x))); }

DEVFN v16bf frag8(uint4 q0, uint4 q1) {
  v8u u;
  u[0] = q0.x; u[1] = q0.y; u[2] = q0.z; u[3] = q0.w;
  u[4] = q1.x; u[5] = q1.y; u[6] = q1.z; u[7] = q1.w;
  return __builtin_bit_cast(v16bf, u);
}
// B fragment: prepacked, lane-contiguous 32 bytes
DEVFN v16bf load_bfrag(const unsigned short* __restrict__ pw, int tile, int lane) {
  const uint4* p = (const uint4*)(pw + ((size_t)(tile * 32 + lane)) * 16);
  return frag8(p[0], p[1]);
}
// A fragment for 16x32 bf16 tile: lane m=l&15 half=l>>4 holds K {0..7,16..23}+half*8
// source points at element k0 = chunk*32 + half*8 within a 64-wide row
DEVFN v16bf load_afrag(const unsigned short* src) {
  const uint4* p = (const uint4*)src;
  return frag8(p[0], p[2]);          // +0 elems and +16 elems
}
DEVFN v8f wmma_bf16(v16bf a, v16bf b, v8f c) {
  return __builtin_amdgcn_wmma_f32_16x16x32_bf16(false, a, false, b, (short)0, c,
                                                 false, false);
}

// ---------------------------------------------------------------------------
// Weight pre-pack into WMMA B-fragment layout (bf16)
//   W: [Kact x 64] f32 row-major.  out tiles: (kt,nt) -> 32 lanes x 16 elems
// ---------------------------------------------------------------------------
__global__ void pack_kernel(const float* __restrict__ W, int Kact, int KT,
                            unsigned short* __restrict__ out) {
  int tid = blockIdx.x * blockDim.x + threadIdx.x;
  int total = KT * 4 * 32;
  if (tid >= total) return;
  int lane = tid & 31, tile = tid >> 5;
  int kt = tile >> 2, nt = tile & 3;
  int lo = lane & 15, half = lane >> 4;
  int n = nt * 16 + lo;
  unsigned short* dst = out + (size_t)tid * 16;
#pragma unroll
  for (int e = 0; e < 16; ++e) {
    int k = kt * 32 + (e & 7) + ((e >> 3) << 4) + half * 8;
    float v = (k < Kact) ? W[(size_t)k * 64 + n] : 0.0f;
    dst[e] = f2bf(v);
  }
}

// ---------------------------------------------------------------------------
// Small utility kernels
// ---------------------------------------------------------------------------
__global__ void zero_kernel(float* __restrict__ p, long n) {
  long i = (long)blockIdx.x * blockDim.x + threadIdx.x;
  if (i < n) p[i] = 0.0f;
}
__global__ void copy_kernel(const float* __restrict__ s, float* __restrict__ d, long n) {
  long i = (long)blockIdx.x * blockDim.x + threadIdx.x;
  if (i < n) d[i] = s[i];
}
// h_emb = h_in @ W[6,64] + b ; also bf16 mirror
__global__ void embed_kernel(const float* __restrict__ hin, const float* __restrict__ w,
                             const float* __restrict__ b, float* __restrict__ hf,
                             unsigned short* __restrict__ hb) {
  long i = (long)blockIdx.x * blockDim.x + threadIdx.x;
  if (i >= (long)NN * 64) return;
  int n = (int)(i >> 6), f = (int)(i & 63);
  const float* hr = hin + (size_t)n * 6;
  float s = b[f];
#pragma unroll
  for (int k = 0; k < 6; ++k) s += hr[k] * w[k * 64 + f];
  hf[i] = s;
  hb[i] = f2bf(s);
}
// radial + normalized diff per edge
__global__ void radial_kernel(const float* __restrict__ xa, const float* __restrict__ xb,
                              const int* __restrict__ rows, const int* __restrict__ cols,
                              float* __restrict__ rad, float* __restrict__ cdf) {
  long e = (long)blockIdx.x * blockDim.x + threadIdx.x;
  if (e >= EE) return;
  int r = rows[e], c = cols[e];
  float d0 = xa[(size_t)r * 3 + 0] - xb[(size_t)c * 3 + 0];
  float d1 = xa[(size_t)r * 3 + 1] - xb[(size_t)c * 3 + 1];
  float d2 = xa[(size_t)r * 3 + 2] - xb[(size_t)c * 3 + 2];
  float rr = d0 * d0 + d1 * d1 + d2 * d2;
  rad[e] = rr;
  float inv = 1.0f / (sqrtf(rr + 1e-8f) + 1.0f);
  cdf[e * 3 + 0] = d0 * inv;
  cdf[e * 3 + 1] = d1 * inv;
  cdf[e * 3 + 2] = d2 * inv;
}
__global__ void cnt_kernel(const int* __restrict__ n2, float* __restrict__ cnt) {
  long e = (long)blockIdx.x * blockDim.x + threadIdx.x;
  if (e < EE) atomicAdd(&cnt[n2[e]], 1.0f);
}
__global__ void noise_h_scatter(const float* __restrict__ h1f, const float* __restrict__ h2f,
                                const int* __restrict__ n1, const int* __restrict__ n2,
                                float* __restrict__ acc) {
  long i = (long)blockIdx.x * blockDim.x + threadIdx.x;
  if (i >= (long)EE * 64) return;
  long j = i >> 6; int f = (int)(i & 63);
  float v = 0.5f * (h1f[(size_t)n1[j] * 64 + f] + h2f[(size_t)n2[j] * 64 + f]);
  atomicAdd(&acc[(size_t)n2[j] * 64 + f], v);
}
__global__ void noise_x_scatter(const float* __restrict__ x1, const float* __restrict__ x2,
                                const int* __restrict__ n1, const int* __restrict__ n2,
                                float* __restrict__ acc) {
  long i = (long)blockIdx.x * blockDim.x + threadIdx.x;
  if (i >= (long)EE * 3) return;
  long j = i / 3; int c = (int)(i % 3);
  float v = 0.5f * (x1[(size_t)n1[j] * 3 + c] + x2[(size_t)n2[j] * 3 + c]);
  atomicAdd(&acc[(size_t)n2[j] * 3 + c], v);
}
__global__ void noise_norm_h(const float* __restrict__ acc, const float* __restrict__ cnt,
                             float* __restrict__ hf, unsigned short* __restrict__ hb) {
  long i = (long)blockIdx.x * blockDim.x + threadIdx.x;
  if (i >= (long)NN * 64) return;
  float v = acc[i] / fmaxf(cnt[i >> 6], 1.0f);
  hf[i] = v;
  hb[i] = f2bf(v);
}
__global__ void noise_norm_x(const float* __restrict__ acc, const float* __restrict__ cnt,
                             float* __restrict__ xc) {
  long i = (long)blockIdx.x * blockDim.x + threadIdx.x;
  if (i >= (long)NN * 3) return;
  xc[i] = acc[i] / fmaxf(cnt[i / 3], 1.0f);
}
// mode 0: x = (x + acc) * nm   mode 1: x = x + fw*acc*nm
__global__ void coord_finish_kernel(float* __restrict__ x, const float* __restrict__ acc,
                                    const float* __restrict__ nm, int mode, float fw) {
  long i = (long)blockIdx.x * blockDim.x + threadIdx.x;
  if (i >= (long)NN * 3) return;
  float m = nm[i / 3];
  x[i] = (mode == 0) ? (x[i] + acc[i]) * m : x[i] + fw * acc[i] * m;
}
__global__ void out_embed_kernel(const float* __restrict__ hf, const float* __restrict__ w,
                                 const float* __restrict__ b, const float* __restrict__ nm,
                                 float* __restrict__ out) {
  long i = (long)blockIdx.x * blockDim.x + threadIdx.x;
  if (i >= (long)NN * 6) return;
  int n = (int)(i / 6), o = (int)(i % 6);
  const float* hr = hf + (size_t)n * 64;
  float s = b[o];
#pragma unroll
  for (int k = 0; k < 64; ++k) s += hr[k] * w[k * 6 + o];
  out[i] = s * nm[n];
}

// ---------------------------------------------------------------------------
// Edge message MLP (WMMA):  m = silu(silu(eh@W1+b1)@W2+b2)*em ; agg[row] += m
//   eh = [h_row(64) | h_col(64) | radial | ea0 | 0-pad] -> K = 160
// One wave per 16-edge tile.
// ---------------------------------------------------------------------------
__global__ __launch_bounds__(32) void edge_mlp_kernel(
    const unsigned short* __restrict__ hRow, const unsigned short* __restrict__ hCol,
    const int* __restrict__ rows, const int* __restrict__ cols,
    const float* __restrict__ rad, const float* __restrict__ ea0,
    const float* __restrict__ em,
    const unsigned short* __restrict__ pW1, const float* __restrict__ b1,
    const unsigned short* __restrict__ pW2, const float* __restrict__ b2,
    float* __restrict__ agg) {
  __shared__ int sRow[16], sCol[16];
  __shared__ float sRad[16], sEa[16], sEm[16];
  __shared__ __align__(16) unsigned short y1s[16 * 64];
  const int l = threadIdx.x;
  const int lo = l & 15, half = l >> 4;
  const size_t e0 = (size_t)blockIdx.x * 16;
  if (l < 16) {
    sRow[l] = rows[e0 + l]; sCol[l] = cols[e0 + l];
    sRad[l] = rad[e0 + l];  sEa[l] = ea0[e0 + l];  sEm[l] = em[e0 + l];
  }
  __syncthreads();

  v8f z = {0, 0, 0, 0, 0, 0, 0, 0};
  v8f acc[4] = {z, z, z, z};
#pragma unroll
  for (int kt = 0; kt < 5; ++kt) {
    v16bf a;
    if (kt < 4) {
      const unsigned short* src =
          (kt < 2) ? hRow + (size_t)sRow[lo] * 64 + kt * 32 + half * 8
                   : hCol + (size_t)sCol[lo] * 64 + (kt - 2) * 32 + half * 8;
      a = load_afrag(src);
    } else {
      v8u u = {0, 0, 0, 0, 0, 0, 0, 0};
      if (half == 0) u[0] = pack2bf(sRad[lo], sEa[lo]);   // k=128 radial, k=129 ea0
      a = __builtin_bit_cast(v16bf, u);
    }
#pragma unroll
    for (int nt = 0; nt < 4; ++nt)
      acc[nt] = wmma_bf16(a, load_bfrag(pW1, kt * 4 + nt, l), acc[nt]);
  }
  // bias + silu, stage bf16 for layer 2 (transpose via LDS)
#pragma unroll
  for (int nt = 0; nt < 4; ++nt) {
    const int n = nt * 16 + lo;
    const float bias = b1[n];
#pragma unroll
    for (int r = 0; r < 8; ++r)
      y1s[(r + 8 * half) * 64 + n] = f2bf(silu_f(acc[nt][r] + bias));
  }
  __syncthreads();

  v8f acc2[4] = {z, z, z, z};
#pragma unroll
  for (int kt = 0; kt < 2; ++kt) {
    v16bf a = load_afrag(y1s + lo * 64 + kt * 32 + half * 8);
#pragma unroll
    for (int nt = 0; nt < 4; ++nt)
      acc2[nt] = wmma_bf16(a, load_bfrag(pW2, kt * 4 + nt, l), acc2[nt]);
  }
#pragma unroll
  for (int nt = 0; nt < 4; ++nt) {
    const int n = nt * 16 + lo;
    const float bias = b2[n];
#pragma unroll
    for (int r = 0; r < 8; ++r) {
      const int m = r + 8 * half;
      const float v = silu_f(acc2[nt][r] + bias) * sEm[m];
      atomicAdd(&agg[(size_t)sRow[m] * 64 + n], v);
    }
  }
}

// ---------------------------------------------------------------------------
// Node MLP (WMMA): y = (concat(h,agg) @ W1 + b1 -> silu) @ W2 + b2
//   mode 0: h = (h + y)*nm       mode 1: h = h + fw*y*nm
// Updates f32 state and bf16 mirror in place (rows owned by this wave).
// ---------------------------------------------------------------------------
__global__ __launch_bounds__(32) void node_mlp_kernel(
    float* __restrict__ hF, unsigned short* __restrict__ hB,
    const float* __restrict__ agg, const float* __restrict__ nm,
    const unsigned short* __restrict__ pW1, const float* __restrict__ b1,
    const unsigned short* __restrict__ pW2, const float* __restrict__ b2,
    int mode, float fw) {
  __shared__ __align__(16) unsigned short y1s[16 * 64];
  const int l = threadIdx.x;
  const int lo = l & 15, half = l >> 4;
  const int base = blockIdx.x * 16;
  v8f z = {0, 0, 0, 0, 0, 0, 0, 0};
  v8f acc[4] = {z, z, z, z};
#pragma unroll
  for (int kt = 0; kt < 4; ++kt) {
    v16bf a;
    if (kt < 2) {
      a = load_afrag(hB + (size_t)(base + lo) * 64 + kt * 32 + half * 8);
    } else {
      const float* f = agg + (size_t)(base + lo) * 64 + (kt - 2) * 32 + half * 8;
      float4 f0 = *(const float4*)(f);
      float4 f1 = *(const float4*)(f + 4);
      float4 g0 = *(const float4*)(f + 16);
      float4 g1 = *(const float4*)(f + 20);
      v8u u;
      u[0] = pack2bf(f0.x, f0.y); u[1] = pack2bf(f0.z, f0.w);
      u[2] = pack2bf(f1.x, f1.y); u[3] = pack2bf(f1.z, f1.w);
      u[4] = pack2bf(g0.x, g0.y); u[5] = pack2bf(g0.z, g0.w);
      u[6] = pack2bf(g1.x, g1.y); u[7] = pack2bf(g1.z, g1.w);
      a = __builtin_bit_cast(v16bf, u);
    }
#pragma unroll
    for (int nt = 0; nt < 4; ++nt)
      acc[nt] = wmma_bf16(a, load_bfrag(pW1, kt * 4 + nt, l), acc[nt]);
  }
#pragma unroll
  for (int nt = 0; nt < 4; ++nt) {
    const int n = nt * 16 + lo;
    const float bias = b1[n];
#pragma unroll
    for (int r = 0; r < 8; ++r)
      y1s[(r + 8 * half) * 64 + n] = f2bf(silu_f(acc[nt][r] + bias));
  }
  __syncthreads();

  v8f acc2[4] = {z, z, z, z};
#pragma unroll
  for (int kt = 0; kt < 2; ++kt) {
    v16bf a = load_afrag(y1s + lo * 64 + kt * 32 + half * 8);
#pragma unroll
    for (int nt = 0; nt < 4; ++nt)
      acc2[nt] = wmma_bf16(a, load_bfrag(pW2, kt * 4 + nt, l), acc2[nt]);
  }
  float nm8[8];
#pragma unroll
  for (int r = 0; r < 8; ++r) nm8[r] = nm[base + r + 8 * half];
#pragma unroll
  for (int nt = 0; nt < 4; ++nt) {
    const int n = nt * 16 + lo;
    const float bias = b2[n];
#pragma unroll
    for (int r = 0; r < 8; ++r) {
      const size_t idx = (size_t)(base + r + 8 * half) * 64 + n;
      const float y = acc2[nt][r] + bias;          // act_last = False
      const float hold = hF[idx];
      const float hnew = (mode == 0) ? (hold + y) * nm8[r] : hold + fw * y * nm8[r];
      hF[idx] = hnew;
      hB[idx] = f2bf(hnew);
    }
  }
}

// ---------------------------------------------------------------------------
// Coord MLP (WMMA + shfl dot): s = silu(eh@CW1+cb1) . cw2 ; xacc[row] += cdiff*s*em
// ---------------------------------------------------------------------------
__global__ __launch_bounds__(32) void coord_mlp_kernel(
    const unsigned short* __restrict__ hRow, const unsigned short* __restrict__ hCol,
    const int* __restrict__ rows, const int* __restrict__ cols,
    const float* __restrict__ rad, const float* __restrict__ ea0,
    const float* __restrict__ cdiff, const float* __restrict__ em,
    const unsigned short* __restrict__ pW1, const float* __restrict__ b1,
    const float* __restrict__ w2, float* __restrict__ xacc) {
  __shared__ int sRow[16], sCol[16];
  __shared__ float sRad[16], sEa[16], sEm[16];
  __shared__ float sDot[16];
  const int l = threadIdx.x;
  const int lo = l & 15, half = l >> 4;
  const size_t e0 = (size_t)blockIdx.x * 16;
  if (l < 16) {
    sRow[l] = rows[e0 + l]; sCol[l] = cols[e0 + l];
    sRad[l] = rad[e0 + l];  sEa[l] = ea0[e0 + l];  sEm[l] = em[e0 + l];
  }
  __syncthreads();

  v8f z = {0, 0, 0, 0, 0, 0, 0, 0};
  v8f acc[4] = {z, z, z, z};
#pragma unroll
  for (int kt = 0; kt < 5; ++kt) {
    v16bf a;
    if (kt < 4) {
      const unsigned short* src =
          (kt < 2) ? hRow + (size_t)sRow[lo] * 64 + kt * 32 + half * 8
                   : hCol + (size_t)sCol[lo] * 64 + (kt - 2) * 32 + half * 8;
      a = load_afrag(src);
    } else {
      v8u u = {0, 0, 0, 0, 0, 0, 0, 0};
      if (half == 0) u[0] = pack2bf(sRad[lo], sEa[lo]);
      a = __builtin_bit_cast(v16bf, u);
    }
#pragma unroll
    for (int nt = 0; nt < 4; ++nt)
      acc[nt] = wmma_bf16(a, load_bfrag(pW1, kt * 4 + nt, l), acc[nt]);
  }
  float t[8] = {0, 0, 0, 0, 0, 0, 0, 0};
#pragma unroll
  for (int nt = 0; nt < 4; ++nt) {
    const int n = nt * 16 + lo;
    const float bias = b1[n];
    const float wv = w2[n];
#pragma unroll
    for (int r = 0; r < 8; ++r) t[r] += silu_f(acc[nt][r] + bias) * wv;
  }
#pragma unroll
  for (int off = 1; off < 16; off <<= 1) {
#pragma unroll
    for (int r = 0; r < 8; ++r) t[r] += __shfl_xor(t[r], off, 32);
  }
  if (lo == 0) {
#pragma unroll
    for (int r = 0; r < 8; ++r) sDot[r + 8 * half] = t[r];
  }
  __syncthreads();
  if (l < 16) {
    const size_t e = e0 + l;
    const float d = sDot[l] * sEm[l];
    const int dst = sRow[l];
#pragma unroll
    for (int c = 0; c < 3; ++c)
      atomicAdd(&xacc[(size_t)dst * 3 + c], cdiff[e * 3 + c] * d);
  }
}

// ---------------------------------------------------------------------------
// Host orchestration
// ---------------------------------------------------------------------------
struct NetP {
  const float *ew1, *eb1, *ew2, *eb2, *nw1, *nb1, *nw2, *nb2, *cw1, *cb1, *cw2;
};

static constexpr size_t PK_EW1 = 0;       // 5 KT -> 10240 us
static constexpr size_t PK_EW2 = 10240;   // 2 KT ->  4096 us
static constexpr size_t PK_NW1 = 14336;   // 4 KT ->  8192 us
static constexpr size_t PK_NW2 = 22528;   // 2 KT ->  4096 us
static constexpr size_t PK_CW1 = 26624;   // 5 KT -> 10240 us
static constexpr size_t PK_STRIDE = 36864;

extern "C" void kernel_launch(void* const* d_in, const int* in_sizes, int n_in,
                              void* d_out, int out_size, void* d_ws, size_t ws_size,
                              hipStream_t stream) {
  (void)out_size; (void)ws_size;
  const float* h1in = (const float*)d_in[0];
  const float* x1in = (const float*)d_in[1];
  const float* h2in = (const float*)d_in[2];
  const float* x2in = (const float*)d_in[3];
  const float* nm1  = (const float*)d_in[4];
  const float* nm2  = (const float*)d_in[5];
  const float* em1  = (const float*)d_in[6];
  const float* em2  = (const float*)d_in[7];
  const float* jem  = (const float*)d_in[8];
  const int* ei1 = (const int*)d_in[9];
  const int* ei2 = (const int*)d_in[10];
  const int* jei = (const int*)d_in[11];
  const int *r1 = ei1, *c1 = ei1 + EE;
  const int *r2 = ei2, *c2 = ei2 + EE;
  const int *jr = jei, *jc = jei + EE;
  const float* emb1w = (const float*)d_in[12];
  const float* emb1b = (const float*)d_in[13];
  const float* emb2w = (const float*)d_in[14];
  const float* emb2b = (const float*)d_in[15];
  const float* embow = (const float*)d_in[16];
  const float* embob = (const float*)d_in[17];

  // --- parameter dict flattening detection --------------------------------
  const bool blobMode = (n_in <= 24);                   // nested dicts as blobs
  const bool alphaMode = !blobMode && (in_sizes[18] == 256);  // tree-sorted keys
  auto getNet = [&](int netIdx) -> NetP {
    NetP p;
    if (blobMode) {
      const float* b = (const float*)d_in[18 + netIdx];
      p.ew1 = b + 0;      p.eb1 = b + 33280;  p.ew2 = b + 33536; p.eb2 = b + 49920;
      p.nw1 = b + 50176;  p.nb1 = b + 82944;  p.nw2 = b + 83200; p.nb2 = b + 99584;
      p.cw1 = b + 99840;  p.cb1 = b + 133120; p.cw2 = b + 133376;
    } else if (alphaMode) {   // cb1,cw1,cw2,eb1,eb2,ew1,ew2,nb1,nb2,nw1,nw2
      int base = 18 + netIdx * 11;
      p.cb1 = (const float*)d_in[base + 0];  p.cw1 = (const float*)d_in[base + 1];
      p.cw2 = (const float*)d_in[base + 2];  p.eb1 = (const float*)d_in[base + 3];
      p.eb2 = (const float*)d_in[base + 4];  p.ew1 = (const float*)d_in[base + 5];
      p.ew2 = (const float*)d_in[base + 6];  p.nb1 = (const float*)d_in[base + 7];
      p.nb2 = (const float*)d_in[base + 8];  p.nw1 = (const float*)d_in[base + 9];
      p.nw2 = (const float*)d_in[base + 10];
    } else {                  // insertion order
      int base = 18 + netIdx * 11;
      p.ew1 = (const float*)d_in[base + 0];  p.eb1 = (const float*)d_in[base + 1];
      p.ew2 = (const float*)d_in[base + 2];  p.eb2 = (const float*)d_in[base + 3];
      p.nw1 = (const float*)d_in[base + 4];  p.nb1 = (const float*)d_in[base + 5];
      p.nw2 = (const float*)d_in[base + 6];  p.nb2 = (const float*)d_in[base + 7];
      p.cw1 = (const float*)d_in[base + 8];  p.cb1 = (const float*)d_in[base + 9];
      p.cw2 = (const float*)d_in[base + 10];
    }
    return p;
  };
  NetP net[3] = {getNet(0), getNet(1), getNet(2)};   // diff, ctrl, fus

  // --- workspace carve -----------------------------------------------------
  char* w = (char*)d_ws;
  auto carve = [&](size_t bytes) -> void* {
    void* p = (void*)w;
    w += (bytes + 255) & ~(size_t)255;
    return p;
  };
  float* h1f = (float*)carve((size_t)NN * 64 * 4);
  float* h2f = (float*)carve((size_t)NN * 64 * 4);
  unsigned short* h1b = (unsigned short*)carve((size_t)NN * 64 * 2);
  unsigned short* h2b = (unsigned short*)carve((size_t)NN * 64 * 2);
  float* x1c = (float*)carve((size_t)NN * 3 * 4);
  float* x2c = (float*)carve((size_t)NN * 3 * 4);
  float* ea1 = (float*)carve((size_t)EE * 4);
  float* ea2 = (float*)carve((size_t)EE * 4);
  float* eaj = (float*)carve((size_t)EE * 4);
  float* crad = (float*)carve((size_t)EE * 4);
  float* cdf  = (float*)carve((size_t)EE * 3 * 4);
  float* agg  = (float*)carve((size_t)NN * 64 * 4);
  float* xac  = (float*)carve((size_t)NN * 3 * 4);
  float* cnt  = (float*)carve((size_t)NN * 4);
  unsigned short* packs = (unsigned short*)carve((size_t)12 * PK_STRIDE * 2);

  auto blks = [](long n, int t) { return dim3((unsigned)((n + t - 1) / t)); };
  auto zerof = [&](float* p, long n) {
    zero_kernel<<<blks(n, 256), 256, 0, stream>>>(p, n);
  };

  // --- pack all weights into WMMA B-fragment layout ------------------------
  for (int g = 0; g < 3; ++g) {
    for (int i = 0; i < LL; ++i) {
      unsigned short* pb = packs + (size_t)(g * 4 + i) * PK_STRIDE;
      pack_kernel<<<blks(5 * 4 * 32, 64), 64, 0, stream>>>(
          net[g].ew1 + (size_t)i * 130 * 64, 130, 5, pb + PK_EW1);
      pack_kernel<<<blks(2 * 4 * 32, 64), 64, 0, stream>>>(
          net[g].ew2 + (size_t)i * 64 * 64, 64, 2, pb + PK_EW2);
      pack_kernel<<<blks(4 * 4 * 32, 64), 64, 0, stream>>>(
          net[g].nw1 + (size_t)i * 128 * 64, 128, 4, pb + PK_NW1);
      pack_kernel<<<blks(2 * 4 * 32, 64), 64, 0, stream>>>(
          net[g].nw2 + (size_t)i * 64 * 64, 64, 2, pb + PK_NW2);
      pack_kernel<<<blks(5 * 4 * 32, 64), 64, 0, stream>>>(
          net[g].cw1 + (size_t)i * 130 * 64, 130, 5, pb + PK_CW1);
    }
  }

  // --- embeddings, initial distances, noise injection ----------------------
  embed_kernel<<<blks((long)NN * 64, 256), 256, 0, stream>>>(h1in, emb1w, emb1b, h1f, h1b);
  embed_kernel<<<blks((long)NN * 64, 256), 256, 0, stream>>>(h2in, emb2w, emb2b, h2f, h2b);
  copy_kernel<<<blks((long)NN * 3, 256), 256, 0, stream>>>(x1in, x1c, (long)NN * 3);
  radial_kernel<<<blks(EE, 256), 256, 0, stream>>>(x1in, x1in, r1, c1, ea1, cdf);
  radial_kernel<<<blks(EE, 256), 256, 0, stream>>>(x2in, x2in, r2, c2, ea2, cdf);
  radial_kernel<<<blks(EE, 256), 256, 0, stream>>>(x1in, x2in, jr, jc, eaj, cdf);

  zerof(cnt, NN);
  zerof(agg, (long)NN * 64);
  zerof(xac, (long)NN * 3);
  cnt_kernel<<<blks(EE, 256), 256, 0, stream>>>(jc, cnt);
  noise_h_scatter<<<blks((long)EE * 64, 256), 256, 0, stream>>>(h1f, h2f, jr, jc, agg);
  noise_x_scatter<<<blks((long)EE * 3, 256), 256, 0, stream>>>(x1in, x2in, jr, jc, xac);
  noise_norm_h<<<blks((long)NN * 64, 256), 256, 0, stream>>>(agg, cnt, h2f, h2b);
  noise_norm_x<<<blks((long)NN * 3, 256), 256, 0, stream>>>(xac, cnt, x2c);

  // --- equivariant block helper --------------------------------------------
  auto runEquiv = [&](int g, int i, float* hf, unsigned short* hb, float* xc,
                      const int* rr, const int* cc, const float* nm,
                      const float* em, const float* ea) {
    const NetP& p = net[g];
    unsigned short* pb = packs + (size_t)(g * 4 + i) * PK_STRIDE;
    radial_kernel<<<blks(EE, 256), 256, 0, stream>>>(xc, xc, rr, cc, crad, cdf);
    zerof(agg, (long)NN * 64);
    edge_mlp_kernel<<<EE / 16, 32, 0, stream>>>(
        hb, hb, rr, cc, crad, ea, em, pb + PK_EW1, p.eb1 + (size_t)i * 64,
        pb + PK_EW2, p.eb2 + (size_t)i * 64, agg);
    node_mlp_kernel<<<NN / 16, 32, 0, stream>>>(
        hf, hb, agg, nm, pb + PK_NW1, p.nb1 + (size_t)i * 64, pb + PK_NW2,
        p.nb2 + (size_t)i * 64, 0, 1.0f);
    zerof(xac, (long)NN * 3);
    coord_mlp_kernel<<<EE / 16, 32, 0, stream>>>(        // uses updated h
        hb, hb, rr, cc, crad, ea, cdf, em, pb + PK_CW1, p.cb1 + (size_t)i * 64,
        p.cw2 + (size_t)i * 64, xac);
    coord_finish_kernel<<<blks((long)NN * 3, 256), 256, 0, stream>>>(xc, xac, nm, 0, 1.0f);
  };

  const float FW = 1.0f;   // FUSION_W[i] == 1.0 for all layers
  for (int i = 0; i < LL; ++i) {
    // control equiv block on graph-2
    runEquiv(1, i, h2f, h2b, x2c, r2, c2, nm2, em2, ea2);

    // fusion block: graph2 -> graph1 (edge/coord use PRE-update h1)
    const NetP& p = net[2];
    unsigned short* pb = packs + (size_t)(2 * 4 + i) * PK_STRIDE;
    radial_kernel<<<blks(EE, 256), 256, 0, stream>>>(x1c, x2c, jr, jc, crad, cdf);
    zerof(agg, (long)NN * 64);
    zerof(xac, (long)NN * 3);
    edge_mlp_kernel<<<EE / 16, 32, 0, stream>>>(
        h1b, h2b, jr, jc, crad, eaj, jem, pb + PK_EW1, p.eb1 + (size_t)i * 64,
        pb + PK_EW2, p.eb2 + (size_t)i * 64, agg);
    coord_mlp_kernel<<<EE / 16, 32, 0, stream>>>(
        h1b, h2b, jr, jc, crad, eaj, cdf, jem, pb + PK_CW1,
        p.cb1 + (size_t)i * 64, p.cw2 + (size_t)i * 64, xac);
    node_mlp_kernel<<<NN / 16, 32, 0, stream>>>(
        h1f, h1b, agg, nm1, pb + PK_NW1, p.nb1 + (size_t)i * 64, pb + PK_NW2,
        p.nb2 + (size_t)i * 64, 1, FW);
    coord_finish_kernel<<<blks((long)NN * 3, 256), 256, 0, stream>>>(x1c, xac, nm1, 1, FW);

    // diffusion equiv block on graph-1
    runEquiv(0, i, h1f, h1b, x1c, r1, c1, nm1, em1, ea1);
  }

  // --- outputs: h1 (N*6) then x1 (N*3) -------------------------------------
  float* out = (float*)d_out;
  out_embed_kernel<<<blks((long)NN * 6, 256), 256, 0, stream>>>(h1f, embow, embob, nm1, out);
  copy_kernel<<<blks((long)NN * 3, 256), 256, 0, stream>>>(x1c, out + (size_t)NN * 6,
                                                           (long)NN * 3);
}
